// Attention_14594298871992
// MI455X (gfx1250) — matrix operand order repaired and checked by reading, not compiled, not run on hardware
//
#include <hip/hip_runtime.h>
#include <hip/hip_bf16.h>

// ---------------------------------------------------------------------------
// GPT attention block with LoRA on Q/V for MI455X (gfx1250, wave32, WMMA).
// All heavy math runs through v_wmma_f32_16x16x32_f16 (f32 accumulate).
// LDS tiles are double-buffered and stored in WMMA *fragment order*
// (frag[lane][16 contiguous f16]) so operand fetch is 2x ds_load_b128 per
// fragment; next-slab global loads are issued before the current slab's
// WMMA chain to hide HBM/L2 latency behind matrix math.
// ---------------------------------------------------------------------------

typedef __attribute__((ext_vector_type(16))) _Float16 v16h;
typedef __attribute__((ext_vector_type(4)))  _Float16 v4h;
typedef __attribute__((ext_vector_type(8)))  float    v8f;

#define BATCH 2
#define SEQ   2048
#define DMODEL 1024
#define NHEAD 16
#define HD    64
#define RANK  16
#define LORA_SCALE 0.5f          // LORA_DIM / LORA_ALPHA = 16/32
#define INV_SQRT_HD 0.125f       // 1/sqrt(64)
#define MROWS 4096               // BATCH*SEQ

__device__ __forceinline__ v8f wmma_f16(v16h a, v16h b, v8f c) {
  return __builtin_amdgcn_wmma_f32_16x16x32_f16(
      /*neg_a=*/false, a, /*neg_b=*/false, b,
      /*c_mod=*/(short)0, c, /*reuse_a=*/false, /*reuse_b=*/false);
}

__device__ __forceinline__ v4h cvt4(float4 f) {
  v4h h;
  h[0] = (_Float16)f.x; h[1] = (_Float16)f.y;
  h[2] = (_Float16)f.z; h[3] = (_Float16)f.w;
  return h;
}

// Fragment-order index helpers (CDNA5 WMMA 16x16x32 f16 layouts):
//   A (16x32): lane = m + 16*((k>>3)&1),  i = ((k>>4)<<3) | (k&7)
//   B (32x16): lane = n + 16*((k>>3)&1),  i = ((k>>4)<<3) | (k&7)
// C/D (16x16 f32): element e, lane L -> row = e + 8*(L>>4), col = L&15.

// A-matrix fragment gathered from global f32 (used for Q; done once).
__device__ __forceinline__ v16h a_frag_gf32(const float* p, int ld) {
  const int lane = threadIdx.x & 31;
  const int m = lane & 15, hh = lane >> 4;
  v16h a;
#pragma unroll
  for (int i = 0; i < 16; ++i) {
    const int k = ((i >> 3) << 4) + (hh << 3) + (i & 7);
    a[i] = (_Float16)p[m * ld + k];
  }
  return a;
}

// ---------------------------------------------------------------------------
// K1: LoRA down-projection. xq = x @ q_a1, xv = x @ v_a1  ([4096,16] each).
// ---------------------------------------------------------------------------
__global__ __launch_bounds__(128) void lora_down_kernel(
    const float* __restrict__ x, const float* __restrict__ qa1,
    const float* __restrict__ va1, float* __restrict__ xq,
    float* __restrict__ xv) {
  const int m = blockIdx.x * 128 + threadIdx.x;      // row in [0,4096)
  const float* xr = x + (size_t)m * DMODEL;
  float aq[RANK], av[RANK];
#pragma unroll
  for (int r = 0; r < RANK; ++r) { aq[r] = 0.f; av[r] = 0.f; }
  for (int k = 0; k < DMODEL; ++k) {
    const float xv_ = xr[k];
#pragma unroll
    for (int r = 0; r < RANK; ++r) {
      aq[r] = fmaf(xv_, qa1[k * RANK + r], aq[r]);
      av[r] = fmaf(xv_, va1[k * RANK + r], av[r]);
    }
  }
#pragma unroll
  for (int r = 0; r < RANK; ++r) {
    xq[(size_t)m * RANK + r] = aq[r];
    xv[(size_t)m * RANK + r] = av[r];
  }
}

// ---------------------------------------------------------------------------
// K2: QKV GEMM (4096 x 3072 x 1024) + bias + LoRA-up + head-major scatter.
// Block = 256 threads (8 waves). Tile: M=128 (16 rows/wave), N=64, K=32.
// Double-buffered LDS; next slab loaded to registers before current WMMAs.
// ---------------------------------------------------------------------------
__global__ __launch_bounds__(256) void qkv_gemm_kernel(
    const float* __restrict__ x, const float* __restrict__ wqkv,
    const float* __restrict__ bqkv, const float* __restrict__ xq,
    const float* __restrict__ xv, const float* __restrict__ qa2,
    const float* __restrict__ va2, float* __restrict__ qbuf,
    float* __restrict__ pk, float* __restrict__ pv) {
  __shared__ __align__(32) _Float16 Asf[2][8][32][16];  // frag order, 2 bufs
  __shared__ __align__(32) _Float16 Bsf[2][4][32][16];
  const int tileN = blockIdx.x * 64;
  const int tileM = blockIdx.y * 128;
  const int tid = threadIdx.x, w = tid >> 5, lane = tid & 31;
  const int hh = lane >> 4, col = lane & 15;

  v8f acc[4];
  const v8f zacc = {};
#pragma unroll
  for (int j = 0; j < 4; ++j) acc[j] = zacc;

  float4 ra[4], rb[2];
  auto load_slab = [&](int kk) {
#pragma unroll
    for (int u = 0; u < 4; ++u) {                   // A: 128x32 slab of x
      const int idx = (tid + u * 256) << 2;
      const int r = idx >> 5, k0 = idx & 31;        // k0 multiple of 4
      ra[u] = *(const float4*)&x[(size_t)(tileM + r) * DMODEL + kk + k0];
    }
#pragma unroll
    for (int u = 0; u < 2; ++u) {                   // B: 32x64 slab of W
      const int idx = (tid + u * 256) << 2;
      const int k = idx >> 6, n0 = idx & 63;        // n0 multiple of 4
      rb[u] = *(const float4*)&wqkv[(size_t)(kk + k) * (3 * DMODEL) +
                                    tileN + n0];
    }
  };
  auto store_slab = [&](int buf) {
#pragma unroll
    for (int u = 0; u < 4; ++u) {
      const int idx = (tid + u * 256) << 2;
      const int r = idx >> 5, k0 = idx & 31;
      const int i0 = ((k0 >> 4) << 3) + (k0 & 7);
      *(v4h*)&Asf[buf][r >> 4][(r & 15) + (((k0 >> 3) & 1) << 4)][i0] =
          cvt4(ra[u]);
    }
#pragma unroll
    for (int u = 0; u < 2; ++u) {
      const int idx = (tid + u * 256) << 2;
      const int k = idx >> 6, n0 = idx & 63;
      const int lhi = ((k >> 3) & 1) << 4, i = ((k >> 4) << 3) | (k & 7), nb = n0 >> 4, nl = n0 & 15;
      Bsf[buf][nb][nl + 0 + lhi][i] = (_Float16)rb[u].x;
      Bsf[buf][nb][nl + 1 + lhi][i] = (_Float16)rb[u].y;
      Bsf[buf][nb][nl + 2 + lhi][i] = (_Float16)rb[u].z;
      Bsf[buf][nb][nl + 3 + lhi][i] = (_Float16)rb[u].w;
    }
  };

  load_slab(0);
  store_slab(0);
  int buf = 0;
  for (int kk = 0; kk < DMODEL; kk += 32) {
    const bool more = (kk + 32) < DMODEL;
    if (more) load_slab(kk + 32);                 // global loads in flight
    __syncthreads();                              // slab `buf` ready
    const v16h a  = *(const v16h*)&Asf[buf][w][lane][0];
    const v16h b0 = *(const v16h*)&Bsf[buf][0][lane][0];
    const v16h b1 = *(const v16h*)&Bsf[buf][1][lane][0];
    const v16h b2 = *(const v16h*)&Bsf[buf][2][lane][0];
    const v16h b3 = *(const v16h*)&Bsf[buf][3][lane][0];
    acc[0] = wmma_f16(a, b0, acc[0]);
    acc[1] = wmma_f16(a, b1, acc[1]);
    acc[2] = wmma_f16(a, b2, acc[2]);
    acc[3] = wmma_f16(a, b3, acc[3]);
    if (more) store_slab(buf ^ 1);                // fill other buffer
    buf ^= 1;
  }

  // Epilogue: bias + LoRA-up + scatter. region uniform per block (64 | 1024).
  const int region = tileN >> 10;                 // 0=q, 1=k, 2=v
#pragma unroll
  for (int j = 0; j < 4; ++j) {
    const int n = tileN + j * 16 + col;           // column in [0,3072)
    const int nn = n - region * DMODEL;           // column within region
    const int h = nn >> 6, d = nn & 63;
    const float bias = bqkv[n];
#pragma unroll
    for (int e = 0; e < 8; ++e) {
      const int gm = tileM + w * 16 + hh * 8 + e; // global row
      float val = acc[j][e] + bias;
      if (region == 0) {
        float s = 0.f;
#pragma unroll
        for (int r = 0; r < RANK; ++r)
          s = fmaf(xq[(size_t)gm * RANK + r], qa2[(size_t)r * DMODEL + nn], s);
        val = fmaf(LORA_SCALE, s, val);
      } else if (region == 2) {
        float s = 0.f;
#pragma unroll
        for (int r = 0; r < RANK; ++r)
          s = fmaf(xv[(size_t)gm * RANK + r], va2[(size_t)r * DMODEL + nn], s);
        val = fmaf(LORA_SCALE, s, val);
      }
      const int bb = gm >> 11, srow = gm & 2047;
      const size_t idx = (((size_t)bb * NHEAD + h) * SEQ + srow) * HD + d;
      if (region == 0)      qbuf[idx] = val;
      else if (region == 1) pk[idx]   = val;
      else                  pv[idx]   = val;
    }
  }
}

// ---------------------------------------------------------------------------
// K3: causal flash attention. Grid (B*H, S/128); block = 256 (8 waves),
// each wave owns 16 query rows. Per 32-key tile: 4 WMMA (QK^T), online
// softmax via 16-lane xor shuffles, P re-swizzled C->A layout through LDS,
// 4 WMMA (P.V). K/V tiles double-buffered and register-pipelined.
// ---------------------------------------------------------------------------
__global__ __launch_bounds__(256) void flash_attn_kernel(
    const float* __restrict__ qbuf, const float* __restrict__ kbuf,
    const float* __restrict__ vbuf, float* __restrict__ amid) {
  // K tile as B-operand of QK^T: [dchunk][keyblk][lane][16], K-dim = d
  __shared__ __align__(32) _Float16 Kbf[2][2][2][32][16];
  // V tile as B-operand of P.V:  [dblk][lane][16], K-dim = key
  __shared__ __align__(32) _Float16 Vbf[2][4][32][16];
  // P round-trip buffer in A-fragment order, per wave
  __shared__ __align__(32) _Float16 Pbf[8][32][16];

  const int bh = blockIdx.x;           // b*NHEAD + h
  const int qbase = blockIdx.y * 128;
  const int tid = threadIdx.x, w = tid >> 5, lane = tid & 31;
  const int hh = lane >> 4, col = lane & 15;

  const float* qrow = qbuf + ((size_t)bh * SEQ + qbase + w * 16) * HD;
  const v16h aq0 = a_frag_gf32(qrow, HD);        // Q[:, 0:32)
  const v16h aq1 = a_frag_gf32(qrow + 32, HD);   // Q[:, 32:64)

  v8f O[4];
  const v8f zacc = {};
#pragma unroll
  for (int j = 0; j < 4; ++j) O[j] = zacc;
  float mrow[8], lrow[8];
#pragma unroll
  for (int e = 0; e < 8; ++e) { mrow[e] = -1e30f; lrow[e] = 0.f; }

  const float* kg = kbuf + (size_t)bh * SEQ * HD;
  const float* vg = vbuf + (size_t)bh * SEQ * HD;
  const int nkt = (qbase + 128) >> 5;            // causal: keys <= qbase+127

  float4 fk[2], fv[2];
  auto load_kv = [&](int k0) {
#pragma unroll
    for (int u = 0; u < 2; ++u) {
      const int idx = (tid + u * 256) << 2;
      const int key = idx >> 6, d0 = idx & 63;    // d0 multiple of 4
      fk[u] = *(const float4*)&kg[(size_t)(k0 + key) * HD + d0];
      fv[u] = *(const float4*)&vg[(size_t)(k0 + key) * HD + d0];
    }
  };
  auto store_kv = [&](int buf) {
#pragma unroll
    for (int u = 0; u < 2; ++u) {
      const int idx = (tid + u * 256) << 2;
      const int key = idx >> 6, d0 = idx & 63;
      // K fragment-order: 4 consecutive d stay in one lane-slot -> b64 store
      const int dc = d0 >> 5;
      const int klane = (key & 15) + (((d0 >> 3) & 1) << 4);
      *(v4h*)&Kbf[buf][dc][key >> 4][klane][(((d0 & 31) >> 4) << 3) | (d0 & 7)] = cvt4(fk[u]);
      // V fragment-order: scatter 4 b16 stores across consecutive lanes
      const int vlhi = ((key >> 3) & 1) << 4, vi = ((key >> 4) << 3) | (key & 7), dl = d0 & 15;
      Vbf[buf][d0 >> 4][dl + 0 + vlhi][vi] = (_Float16)fv[u].x;
      Vbf[buf][d0 >> 4][dl + 1 + vlhi][vi] = (_Float16)fv[u].y;
      Vbf[buf][d0 >> 4][dl + 2 + vlhi][vi] = (_Float16)fv[u].z;
      Vbf[buf][d0 >> 4][dl + 3 + vlhi][vi] = (_Float16)fv[u].w;
    }
  };

  load_kv(0);
  store_kv(0);
  int buf = 0;
  for (int kt = 0; kt < nkt; ++kt) {
    const int k0 = kt * 32;
    const bool more = (kt + 1) < nkt;
    if (more) load_kv(k0 + 32);                   // global loads in flight
    __syncthreads();                              // tile `buf` ready

    // scores: 16 queries x 32 keys, K-dim = hd = 64 -> 2 WMMA per 16 keys
    const v16h kb00 = *(const v16h*)&Kbf[buf][0][0][lane][0];
    const v16h kb10 = *(const v16h*)&Kbf[buf][1][0][lane][0];
    const v16h kb01 = *(const v16h*)&Kbf[buf][0][1][lane][0];
    const v16h kb11 = *(const v16h*)&Kbf[buf][1][1][lane][0];
    v8f s0 = zacc, s1 = zacc;
    s0 = wmma_f16(aq0, kb00, s0);
    s0 = wmma_f16(aq1, kb10, s0);
    s1 = wmma_f16(aq0, kb01, s1);
    s1 = wmma_f16(aq1, kb11, s1);

    float p0[8], p1[8];
#pragma unroll
    for (int e = 0; e < 8; ++e) {
      const int qr = qbase + w * 16 + hh * 8 + e;   // global query row
      float x0 = s0[e] * INV_SQRT_HD;
      float x1 = s1[e] * INV_SQRT_HD;
      if (k0 + col > qr)      x0 = -1e30f;          // causal mask
      if (k0 + 16 + col > qr) x1 = -1e30f;
      float tmax = fmaxf(x0, x1);
#pragma unroll
      for (int msk = 1; msk < 16; msk <<= 1)        // row max (16-lane half)
        tmax = fmaxf(tmax, __shfl_xor(tmax, msk, 32));
      const float mnew = fmaxf(mrow[e], tmax);
      const float fac = __expf(mrow[e] - mnew);
      const float e0 = __expf(x0 - mnew);
      const float e1 = __expf(x1 - mnew);
      float rsum = e0 + e1;
#pragma unroll
      for (int msk = 1; msk < 16; msk <<= 1)        // row sum
        rsum += __shfl_xor(rsum, msk, 32);
      lrow[e] = lrow[e] * fac + rsum;
      mrow[e] = mnew;
      p0[e] = e0; p1[e] = e1;
#pragma unroll
      for (int j = 0; j < 4; ++j) O[j][e] *= fac;   // rescale accumulator
    }

    // C-layout -> A-fragment-order for P (this wave's private LDS slice):
    // P[row, key]: dest lane = row + 16*((key>>3)&1), i = ((key>>4)<<3)|(key&7)
#pragma unroll
    for (int e = 0; e < 8; ++e) {
      const int row = hh * 8 + e;
      const int plane = row + ((col >> 3) << 4);
      Pbf[w][plane][col & 7]       = (_Float16)p0[e];   // key = col
      Pbf[w][plane][8 + (col & 7)] = (_Float16)p1[e];   // key = 16 + col
    }
    const v16h ap = *(const v16h*)&Pbf[w][lane][0];     // in-wave DS ordering
    const v16h vb0 = *(const v16h*)&Vbf[buf][0][lane][0];
    const v16h vb1 = *(const v16h*)&Vbf[buf][1][lane][0];
    const v16h vb2 = *(const v16h*)&Vbf[buf][2][lane][0];
    const v16h vb3 = *(const v16h*)&Vbf[buf][3][lane][0];
    O[0] = wmma_f16(ap, vb0, O[0]);
    O[1] = wmma_f16(ap, vb1, O[1]);
    O[2] = wmma_f16(ap, vb2, O[2]);
    O[3] = wmma_f16(ap, vb3, O[3]);

    if (more) store_kv(buf ^ 1);                  // fill other buffer
    buf ^= 1;
  }

  // normalize and write merged-head activation [B,S,D]
  const int bb = bh >> 4, h = bh & 15;
#pragma unroll
  for (int e = 0; e < 8; ++e) {
    const float inv = 1.0f / fmaxf(lrow[e], 1e-20f);
    const int srow = qbase + w * 16 + hh * 8 + e;
    const size_t base = (((size_t)bb * SEQ + srow) * NHEAD + h) * HD;
#pragma unroll
    for (int j = 0; j < 4; ++j) amid[base + j * 16 + col] = O[j][e] * inv;
  }
}

// ---------------------------------------------------------------------------
// K4: output projection 4096 x 1024 x 1024 + bias -> d_out[0 : B*S*D)
// ---------------------------------------------------------------------------
__global__ __launch_bounds__(256) void proj_gemm_kernel(
    const float* __restrict__ amid, const float* __restrict__ wproj,
    const float* __restrict__ bproj, float* __restrict__ out) {
  __shared__ __align__(32) _Float16 Asf[2][8][32][16];
  __shared__ __align__(32) _Float16 Bsf[2][4][32][16];
  const int tileN = blockIdx.x * 64;
  const int tileM = blockIdx.y * 128;
  const int tid = threadIdx.x, w = tid >> 5, lane = tid & 31;
  const int hh = lane >> 4, col = lane & 15;

  v8f acc[4];
  const v8f zacc = {};
#pragma unroll
  for (int j = 0; j < 4; ++j) acc[j] = zacc;

  float4 ra[4], rb[2];
  auto load_slab = [&](int kk) {
#pragma unroll
    for (int u = 0; u < 4; ++u) {
      const int idx = (tid + u * 256) << 2;
      const int r = idx >> 5, k0 = idx & 31;
      ra[u] = *(const float4*)&amid[(size_t)(tileM + r) * DMODEL + kk + k0];
    }
#pragma unroll
    for (int u = 0; u < 2; ++u) {
      const int idx = (tid + u * 256) << 2;
      const int k = idx >> 6, n0 = idx & 63;
      rb[u] = *(const float4*)&wproj[(size_t)(kk + k) * DMODEL + tileN + n0];
    }
  };
  auto store_slab = [&](int buf) {
#pragma unroll
    for (int u = 0; u < 4; ++u) {
      const int idx = (tid + u * 256) << 2;
      const int r = idx >> 5, k0 = idx & 31;
      const int i0 = ((k0 >> 4) << 3) + (k0 & 7);
      *(v4h*)&Asf[buf][r >> 4][(r & 15) + (((k0 >> 3) & 1) << 4)][i0] =
          cvt4(ra[u]);
    }
#pragma unroll
    for (int u = 0; u < 2; ++u) {
      const int idx = (tid + u * 256) << 2;
      const int k = idx >> 6, n0 = idx & 63;
      const int lhi = ((k >> 3) & 1) << 4, i = ((k >> 4) << 3) | (k & 7), nb = n0 >> 4, nl = n0 & 15;
      Bsf[buf][nb][nl + 0 + lhi][i] = (_Float16)rb[u].x;
      Bsf[buf][nb][nl + 1 + lhi][i] = (_Float16)rb[u].y;
      Bsf[buf][nb][nl + 2 + lhi][i] = (_Float16)rb[u].z;
      Bsf[buf][nb][nl + 3 + lhi][i] = (_Float16)rb[u].w;
    }
  };

  load_slab(0);
  store_slab(0);
  int buf = 0;
  for (int kk = 0; kk < DMODEL; kk += 32) {
    const bool more = (kk + 32) < DMODEL;
    if (more) load_slab(kk + 32);
    __syncthreads();
    const v16h a  = *(const v16h*)&Asf[buf][w][lane][0];
    const v16h b0 = *(const v16h*)&Bsf[buf][0][lane][0];
    const v16h b1 = *(const v16h*)&Bsf[buf][1][lane][0];
    const v16h b2 = *(const v16h*)&Bsf[buf][2][lane][0];
    const v16h b3 = *(const v16h*)&Bsf[buf][3][lane][0];
    acc[0] = wmma_f16(a, b0, acc[0]);
    acc[1] = wmma_f16(a, b1, acc[1]);
    acc[2] = wmma_f16(a, b2, acc[2]);
    acc[3] = wmma_f16(a, b3, acc[3]);
    if (more) store_slab(buf ^ 1);
    buf ^= 1;
  }

#pragma unroll
  for (int j = 0; j < 4; ++j) {
    const int n = tileN + j * 16 + col;
    const float bias = bproj[n];
#pragma unroll
    for (int e = 0; e < 8; ++e) {
      const int gm = tileM + w * 16 + hh * 8 + e;
      out[(size_t)gm * DMODEL + n] = acc[j][e] + bias;
    }
  }
}

// ---------------------------------------------------------------------------
extern "C" void kernel_launch(void* const* d_in, const int* in_sizes, int n_in,
                              void* d_out, int out_size, void* d_ws, size_t ws_size,
                              hipStream_t stream) {
  const float* x        = (const float*)d_in[0];   // [B,S,D]
  const float* c_attn_w = (const float*)d_in[1];   // [D,3D]
  const float* c_attn_b = (const float*)d_in[2];   // [3D]
  const float* c_proj_w = (const float*)d_in[3];   // [D,D]
  const float* c_proj_b = (const float*)d_in[4];   // [D]
  const float* q_a1     = (const float*)d_in[5];   // [D,16]
  const float* q_a2     = (const float*)d_in[6];   // [16,D]
  const float* v_a1     = (const float*)d_in[7];   // [D,16]
  const float* v_a2     = (const float*)d_in[8];   // [16,D]

  float* out_a = (float*)d_out;                                  // [B,S,D]
  float* pk    = out_a + (size_t)MROWS * DMODEL;                 // present[0]=kh
  float* pv    = pk + (size_t)BATCH * NHEAD * SEQ * HD;          // present[1]=vh

  float* ws   = (float*)d_ws;
  float* xq   = ws;                                              // [4096,16]
  float* xv   = xq + (size_t)MROWS * RANK;                       // [4096,16]
  float* qbuf = xv + (size_t)MROWS * RANK;                       // [B,H,S,hd]
  float* amid = qbuf + (size_t)BATCH * NHEAD * SEQ * HD;         // [B,S,D]

  lora_down_kernel<<<dim3(MROWS / 128), dim3(128), 0, stream>>>(
      x, q_a1, v_a1, xq, xv);

  qkv_gemm_kernel<<<dim3(3 * DMODEL / 64, MROWS / 128), dim3(256), 0, stream>>>(
      x, c_attn_w, c_attn_b, xq, xv, q_a2, v_a2, qbuf, pk, pv);

  flash_attn_kernel<<<dim3(BATCH * NHEAD, SEQ / 128), dim3(256), 0, stream>>>(
      qbuf, pk, pv, amid);

  proj_gemm_kernel<<<dim3(DMODEL / 64, MROWS / 128), dim3(256), 0, stream>>>(
      amid, c_proj_w, c_proj_b, out_a);
}
